// EncoderOnlyModel_4380866642059
// MI455X (gfx1250) — matrix operand order repaired and checked by reading, not compiled, not run on hardware
//
#include <hip/hip_runtime.h>
#include <hip/hip_bf16.h>

typedef __attribute__((ext_vector_type(16))) _Float16 v16h;
typedef __attribute__((ext_vector_type(8)))  float    v8f;
typedef __attribute__((ext_vector_type(4)))  float    f4;

#define OUT_CH 4
#define C_DIM 64
#define D_GRID 32
#define K_NBR 8
#define B_GR 4
#define M_TOK (D_GRID * D_GRID * D_GRID)          /* 32768 */
#define TOTAL_TOK (B_GR * M_TOK)                  /* 131072 */
#define PAIRS (TOTAL_TOK / 2)                     /* 65536 */
#define PAIRS_PER_WAVE 16
#define WAVES_PER_BLOCK 8
#define BLOCKS_MAIN (PAIRS / (PAIRS_PER_WAVE * WAVES_PER_BLOCK)) /* 512 */

struct F3 { float a, b, c; };   // 12-byte row -> wide global load

// jax.nn.gelu default (approximate=True, tanh form), native v_tanh_f32 when
// available; else exp2/rcp sigmoid identity 0.5*(1+tanh(u)) == sigmoid(2u).
__device__ __forceinline__ float fast_gelu(float v) {
    const float u = 0.7978845608028654f * v * __builtin_fmaf(0.044715f, v * v, 1.0f);
#if __has_builtin(__builtin_amdgcn_tanhf)
    return 0.5f * v * (1.0f + __builtin_amdgcn_tanhf(u));
#elif __has_builtin(__builtin_amdgcn_exp2f) && __has_builtin(__builtin_amdgcn_rcpf)
    const float e = __builtin_amdgcn_exp2f(-2.8853900817779268f * u);
    return v * __builtin_amdgcn_rcpf(1.0f + e);
#else
    return 0.5f * v * (1.0f + tanhf(u));
#endif
}

// Compute 8 contiguous hidden channels [base..base+7]:
// gelu(rel . W_k1[:,c] + b_k1[c]) -> f16
__device__ __forceinline__ void hid8(const float* sW, const float* sB, int base,
                                     float r0, float r1, float r2, _Float16* dst) {
    float bb[8], w0[8], w1[8], w2[8];
    *(f4*)&bb[0] = *(const f4*)&sB[base];
    *(f4*)&bb[4] = *(const f4*)&sB[base + 4];
    *(f4*)&w0[0] = *(const f4*)&sW[base];
    *(f4*)&w0[4] = *(const f4*)&sW[base + 4];
    *(f4*)&w1[0] = *(const f4*)&sW[64 + base];
    *(f4*)&w1[4] = *(const f4*)&sW[64 + base + 4];
    *(f4*)&w2[0] = *(const f4*)&sW[128 + base];
    *(f4*)&w2[4] = *(const f4*)&sW[128 + base + 4];
#pragma unroll
    for (int i = 0; i < 8; ++i) {
        float h = bb[i] + r0 * w0[i] + r1 * w1[i] + r2 * w2[i];
        dst[i]  = (_Float16)fast_gelu(h);
    }
}

__global__ void zero_pooled_kernel(float* ws) {
    ws[threadIdx.x] = 0.0f; // 256 floats = pooled[B_GR][C_DIM]
}

__global__ __launch_bounds__(256, 4) void gno_main_kernel(
    const float* __restrict__ pos, const float* __restrict__ x,
    const float* __restrict__ W_lift, const float* __restrict__ b_lift,
    const float* __restrict__ W_k1, const float* __restrict__ b_k1,
    const float* __restrict__ W_k2, const float* __restrict__ b_k2,
    const int* __restrict__ nbr, float* __restrict__ pooled)
{
    __shared__ float    sWk1[192];
    __shared__ float    sbk1[64];
    __shared__ _Float16 sWk2h[8 * 32 * 16]; // [s*4+j][lane][e], f16 B-fragments (8 KB)

    const int tid = threadIdx.x;
    if (tid < 192) sWk1[tid] = W_k1[tid];
    else           sbk1[tid - 192] = b_k1[tid - 192];

    // Pre-convert W_k2 into per-lane WMMA B-fragment layout in LDS.
    {
        const int sj    = tid >> 5;          // 0..7 = s*4 + j
        const int laneq = tid & 31;
        const int s     = sj >> 2;
        const int j     = sj & 3;
        const int hiBq  = (laneq & 16) ? 8 : 0;
        const int cq    = 16 * j + (laneq & 15);
        const int base  = tid * 16;
#pragma unroll
        for (int e = 0; e < 16; ++e) {
            const int kk = 32 * s + hiBq + (e < 8 ? e : e + 8); // hidden-channel row of W_k2
            sWk2h[base + e] = (_Float16)W_k2[kk * C_DIM + cq];
        }
    }
    __syncthreads();

    const int lane  = tid & 31;
    const int wave  = tid >> 5;
    const int gwave = blockIdx.x * WAVES_PER_BLOCK + wave;
    const int pair0 = gwave * PAIRS_PER_WAVE;
    const int c15   = lane & 15;
    const int hiA   = (lane & 16) ? 8 : 0;   // A-fragment channel offset
    const int Mr    = lane & 15;             // A-matrix row this lane feeds

    const F3*   pos3 = (const F3*)pos;
    const F3*   x3   = (const F3*)x;
    const int4* nbr4 = (const int4*)nbr;     // K_NBR=8 -> 2 int4 per token

    // LDS byte address of this lane's B-fragment for n-tile j (s=0 plane);
    // s=1 plane is at a constant +4096 bytes.  (flat LDS aperture: addr[31:0]
    // is the LDS byte offset)
    const unsigned ldsB = (unsigned)(size_t)&sWk2h[0] + (unsigned)lane * 32u;

    // Hoisted per-lane column constants (output channels c = 16*j + c15)
    float blf[4], bk2v[4], wl0[4], wl1[4], wl2[4];
#pragma unroll
    for (int j = 0; j < 4; ++j) {
        const int c = 16 * j + c15;
        blf[j]  = b_lift[c];
        bk2v[j] = b_k2[c];
        wl0[j]  = W_lift[0 * C_DIM + c];
        wl1[j]  = W_lift[1 * C_DIM + c];
        wl2[j]  = W_lift[2 * C_DIM + c];
    }

    float tok[4] = {0.f, 0.f, 0.f, 0.f};

    union BFrag { v16h v; f4 q[2]; };

#pragma unroll 1
    for (int p = 0; p < PAIRS_PER_WAVE; ++p) {
        const int g0 = (pair0 + p) * 2; // global token index of first token in pair

        // ---- A side: build gelu(rel @ W_k1 + b_k1) fragment for row Mr ----
        const int tA = g0 + (Mr >> 3);
        const unsigned nA = (unsigned)nbr[8u * (unsigned)tA + (unsigned)(Mr & 7)];
        const F3 pp = pos3[nA];
        const int m = tA & (M_TOK - 1);
        const float lx = (float)(m >> 10)       * (2.0f / 31.0f) - 1.0f;
        const float ly = (float)((m >> 5) & 31) * (2.0f / 31.0f) - 1.0f;
        const float lz = (float)(m & 31)        * (2.0f / 31.0f) - 1.0f;
        const float r0 = pp.a - lx, r1 = pp.b - ly, r2 = pp.c - lz;

        v16h afr[2];
#pragma unroll
        for (int s = 0; s < 2; ++s) {
            _Float16 tmp[16];
            hid8(sWk1, sbk1, 32 * s + hiA,      r0, r1, r2, &tmp[0]);
            hid8(sWk1, sbk1, 32 * s + hiA + 16, r0, r1, r2, &tmp[8]);
#pragma unroll
            for (int e = 0; e < 16; ++e) afr[s][e] = tmp[e];
        }

        // ---- C-side gathers (reused by all 4 n-tiles) ----
        const unsigned tC = (unsigned)(g0 + (lane >> 4)); // lanes 0-15: g0, 16-31: g0+1
        const int4 i0 = nbr4[2u * tC + 0u];
        const int4 i1 = nbr4[2u * tC + 1u];
        int idx[8] = { i0.x, i0.y, i0.z, i0.w, i1.x, i1.y, i1.z, i1.w };
        float hx[8], hy[8], hz[8];
#pragma unroll
        for (int k = 0; k < 8; ++k) {
            const F3 xv = x3[(unsigned)idx[k]];
            hx[k] = xv.a; hy[k] = xv.b; hz[k] = xv.c;
        }

        // ---- Per n-tile: B-frag via explicit ds_load_b128 (un-hoistable),
        //      2x WMMA, fused epilogue ----
#pragma unroll
        for (int j = 0; j < 4; ++j) {
            BFrag b0, b1;
            const unsigned a0 = ldsB + (unsigned)j * 1024u; // (j*32 lanes)*32 B
            asm volatile(
                "ds_load_b128 %0, %4\n\t"
                "ds_load_b128 %1, %4 offset:16\n\t"
                "ds_load_b128 %2, %4 offset:4096\n\t"
                "ds_load_b128 %3, %4 offset:4112\n\t"
                "s_wait_dscnt 0x0"
                : "=&v"(b0.q[0]), "=&v"(b0.q[1]), "=&v"(b1.q[0]), "=&v"(b1.q[1])
                : "v"(a0));

            v8f z = {};
            z = __builtin_amdgcn_wmma_f32_16x16x32_f16(false, afr[0], false, b0.v,
                                                       (short)0, z, false, false);
            z = __builtin_amdgcn_wmma_f32_16x16x32_f16(false, afr[1], false, b1.v,
                                                       (short)0, z, false, false);

            // msg = (ker + b_k2) * (x[nbr] @ W_lift + b_lift), summed over K
            float sum = 0.f;
#pragma unroll
            for (int r = 0; r < 8; ++r) {
                const float hval = blf[j] + hx[r] * wl0[j] + hy[r] * wl1[j] + hz[r] * wl2[j];
                sum += (z[r] + bk2v[j]) * hval;
            }
            tok[j] += sum;
        }
    }

    // Per-graph accumulation (wave's 32-token chunk never crosses a graph)
    const int b = (pair0 * 2) >> 15; // / M_TOK
#pragma unroll
    for (int j = 0; j < 4; ++j)
        atomicAdd(&pooled[b * C_DIM + 16 * j + c15], tok[j]);
}

__global__ void finalize_kernel(const float* __restrict__ pooled,
                                const float* __restrict__ W_out,
                                const float* __restrict__ b_out,
                                float* __restrict__ out4) {
    const int t = threadIdx.x;
    if (t < B_GR * OUT_CH) {
        const int b = t >> 2, o = t & 3;
        float s = 0.f;
        for (int c = 0; c < C_DIM; ++c)
            s += pooled[b * C_DIM + c] * W_out[c * OUT_CH + o];
        out4[t] = s * (1.0f / (float)(M_TOK * K_NBR)) + b_out[o];
    }
}

__global__ void broadcast_kernel(const int* __restrict__ batch_idx,
                                 const float* __restrict__ out4,
                                 float* __restrict__ out, int N) {
    const int n = blockIdx.x * blockDim.x + threadIdx.x;
    if (n < N) {
        const int b = batch_idx[n];
        ((float4*)out)[n] = ((const float4*)out4)[b];
    }
}

extern "C" void kernel_launch(void* const* d_in, const int* in_sizes, int n_in,
                              void* d_out, int out_size, void* d_ws, size_t ws_size,
                              hipStream_t stream) {
    const float* pos       = (const float*)d_in[0];
    const float* x         = (const float*)d_in[1];
    const float* W_lift    = (const float*)d_in[2];
    const float* b_lift    = (const float*)d_in[3];
    const float* W_k1      = (const float*)d_in[4];
    const float* b_k1      = (const float*)d_in[5];
    const float* W_k2      = (const float*)d_in[6];
    const float* b_k2      = (const float*)d_in[7];
    const float* W_out     = (const float*)d_in[8];
    const float* b_out     = (const float*)d_in[9];
    const int*   batch_idx = (const int*)d_in[10];
    const int*   nbr_idx   = (const int*)d_in[11];

    float* ws     = (float*)d_ws;
    float* pooled = ws;        // B_GR*C_DIM = 256 floats
    float* out4   = ws + 256;  // B_GR*OUT_CH = 16 floats
    const int N   = in_sizes[0] / 3;

    zero_pooled_kernel<<<1, 256, 0, stream>>>(pooled);
    gno_main_kernel<<<BLOCKS_MAIN, 256, 0, stream>>>(
        pos, x, W_lift, b_lift, W_k1, b_k1, W_k2, b_k2, nbr_idx, pooled);
    finalize_kernel<<<1, 64, 0, stream>>>(pooled, W_out, b_out, out4);
    broadcast_kernel<<<(N + 255) / 256, 256, 0, stream>>>(batch_idx, out4,
                                                          (float*)d_out, N);
}
